// JointNet_15650860827001
// MI455X (gfx1250) — matrix operand order repaired and checked by reading, not compiled, not run on hardware
//
#include <hip/hip_runtime.h>
#include <hip/hip_bf16.h>

// ---------------------------------------------------------------------------
// RNN-T joint network for gfx1250 (MI455X), wave32 + v_wmma_f32_16x16x32_f16.
//
// Shapes: T=200, U=50, N=8, ENC=PRED=512, H=640, V=1000.
//   proj GEMMs: (2000 x 512) x (512 x 640)        -> f32 proj in ws
//   main GEMM : M=80000 rows of tanh(e+p+b1) (f16, built in LDS, BM=64)
//               x packed W2 (K=640, V=1000, f16)  -> f32 out (N,T,U,V)
// ---------------------------------------------------------------------------

typedef __attribute__((ext_vector_type(16))) _Float16 v16h;
typedef __attribute__((ext_vector_type(8)))  _Float16 v8h;
typedef __attribute__((ext_vector_type(4)))  _Float16 v4h;
typedef __attribute__((ext_vector_type(8)))  float    v8f;

#define SHUF16(lo, hi) __builtin_shufflevector((lo), (hi), 0,1,2,3,4,5,6,7,8,9,10,11,12,13,14,15)

// Hardware transcendental tanh if the builtin exists on this toolchain.
#if __has_builtin(__builtin_amdgcn_tanhf)
  #define TANH_F32(x) __builtin_amdgcn_tanhf(x)
#elif __has_builtin(__builtin_amdgcn_tanh_f32)
  #define TANH_F32(x) __builtin_amdgcn_tanh_f32(x)
#else
  #define TANH_F32(x) tanhf(x)
#endif

// ---- workspace layout (bytes), all 256B aligned -----------------------------
// Xh    : 2000*512 f16            = 2,048,000
// W1PE  : 16kt*40nt*32*16 f16     =   655,360
// W1PP  : same                    =   655,360
// W2P   : 20kt*63nt*32*16 f16     = 1,290,240
// encP  : 1600*640 f32            = 4,096,000
// predP :  400*640 f32            = 1,024,000   (total ~9.77 MB)
#define OFF_XH    0
#define OFF_W1PE  2048000
#define OFF_W1PP  2703360
#define OFF_W2P   3358720
#define OFF_ENCP  4648960
#define OFF_PREDP 8744960

// ---------------------------------------------------------------------------
// Convert enc_state (1600x512) + pred_state (400x512) f32 -> Xh f16 (2000x512)
__global__ __launch_bounds__(256) void k_cvtX(const float* __restrict__ enc,
                                              const float* __restrict__ pred,
                                              _Float16* __restrict__ dst) {
  int idx = blockIdx.x * 256 + threadIdx.x;           // < 1,024,000
  if (idx >= 2000 * 512) return;
  float v = (idx < 1600 * 512) ? enc[idx] : pred[idx - 1600 * 512];
  dst[idx] = (_Float16)v;
}

// ---------------------------------------------------------------------------
// Pack a weight matrix into WMMA B-fragment-major f16 layout:
//   dst[((kt*nTiles + nt)*32 + lane)*16 + e] = W[row][dOff + k]
//   row = nt*16 + (lane&15),  k = kt*32 + (lane>=16 ? 8 : 0) + e + (e>=8 ? 8 : 0)
// (B 32x16 f16 layout: VGPR i<4 holds K=2i,2i+1 (lanes 0-15) / K=8+2i,.. (16-31); VGPR i+4 adds 16)
__global__ __launch_bounds__(256) void k_packW(const float* __restrict__ W,
                                               int rowPitch, int dOff,
                                               int nTiles, int rowMax, int kTiles,
                                               _Float16* __restrict__ dst) {
  int idx = blockIdx.x * 256 + threadIdx.x;
  if (idx >= kTiles * nTiles * 512) return;
  int e    = idx & 15;
  int lane = (idx >> 4) & 31;
  int tile = idx >> 9;
  int nt   = tile % nTiles;
  int kt   = tile / nTiles;
  int row  = nt * 16 + (lane & 15);
  int k    = kt * 32 + (lane >> 4) * 8 + e + ((e >> 3) << 3);
  float val = (row < rowMax) ? W[(size_t)row * rowPitch + dOff + k] : 0.0f;
  dst[idx] = (_Float16)val;
}

// ---------------------------------------------------------------------------
// Projections via WMMA: 125 M-tiles (0-99 enc, 100-124 pred) x 40 N-tiles,
// one 16x16 tile per wave, K=512 in 16 steps of 32.
__global__ __launch_bounds__(256) void k_proj(const _Float16* __restrict__ Xh,
                                              const _Float16* __restrict__ W1PE,
                                              const _Float16* __restrict__ W1PP,
                                              float* __restrict__ encP,
                                              float* __restrict__ predP) {
  int tid  = threadIdx.x;
  int wave = tid >> 5, lane = tid & 31;
  int tileId = blockIdx.x * 8 + wave;                 // < 5000
  int mt = tileId / 40, nt = tileId % 40;
  int laneHi = lane >> 4, ln = lane & 15;
  int s = laneHi * 8;

  const _Float16* AP = Xh + (size_t)(mt * 16 + ln) * 512;
  const _Float16* BP = ((mt >= 100) ? W1PP : W1PE) + ((size_t)nt * 32 + lane) * 16;

  v8f acc = {};
  for (int kt = 0; kt < 16; ++kt) {
    int k0 = kt * 32;
    v8h alo = *(const v8h*)(AP + k0 + s);
    v8h ahi = *(const v8h*)(AP + k0 + s + 16);
    v8h blo = *(const v8h*)(BP);
    v8h bhi = *(const v8h*)(BP + 8);
    BP += 40 * 512;
    v16h A = SHUF16(alo, ahi);
    v16h B = SHUF16(blo, bhi);
    acc = __builtin_amdgcn_wmma_f32_16x16x32_f16(false, A, false, B,
                                                 (short)0, acc, false, false);
  }
  int h = nt * 16 + ln;
  union { v8f v; float f[8]; } au; au.v = acc;
  for (int j = 0; j < 8; ++j) {
    int r = mt * 16 + j + laneHi * 8;
    if (r < 1600) encP[(size_t)r * 640 + h] = au.f[j];
    else          predP[(size_t)(r - 1600) * 640 + h] = au.f[j];
  }
}

// ---------------------------------------------------------------------------
// Fused main kernel. Each block: 64 rows of M=80000 (m=(t*50+u)*8+n).
// Phase 1: hidden tile tanh(e+p+b1) -> LDS f16 [64][648] (padded pitch,
//          648*2 = 1296 B = 324 dwords, 324%64=4 -> conflict-free rows).
// Phase 2: 8 waves sweep 63 V-tiles; per tile, each B fragment is loaded once
//          and reused by FOUR 16-row M-subtiles (4 accumulators, 20 k-steps).
//          Halves W2 L2 traffic vs BM=32 (~1.6 GB total for the dispatch).
__global__ __launch_bounds__(256) void k_joint(const float* __restrict__ encP,
                                               const float* __restrict__ predP,
                                               const float* __restrict__ b1,
                                               const _Float16* __restrict__ W2P,
                                               const float* __restrict__ b2,
                                               float* __restrict__ out) {
  __shared__ _Float16 sH[64 * 648];                  // 82,944 B (<320KB/WGP)
  int tid = threadIdx.x;
  int m0  = blockIdx.x * 64;

  // ---- phase 1: build hidden tile --------------------------------------
  for (int it = tid; it < 64 * 160; it += 256) {     // 160 float4 units / row
    int r  = it / 160;
    int c4 = it - r * 160;
    int h0 = c4 * 4;
    int m  = m0 + r;
    int t  = m / 400;
    int rem = m - t * 400;
    int u = rem >> 3, n = rem & 7;
    float4 e4 = *(const float4*)(encP  + (size_t)(t * 8 + n) * 640 + h0);
    float4 p4 = *(const float4*)(predP + (size_t)(u * 8 + n) * 640 + h0);
    float4 b4 = *(const float4*)(b1 + h0);
    v4h hv;
    hv[0] = (_Float16)TANH_F32(e4.x + p4.x + b4.x);
    hv[1] = (_Float16)TANH_F32(e4.y + p4.y + b4.y);
    hv[2] = (_Float16)TANH_F32(e4.z + p4.z + b4.z);
    hv[3] = (_Float16)TANH_F32(e4.w + p4.w + b4.w);
    *(v4h*)(&sH[r * 648 + h0]) = hv;
  }
  __syncthreads();

  // ---- phase 2: WMMA over V tiles --------------------------------------
  int wave = tid >> 5, lane = tid & 31;
  int laneHi = lane >> 4, ln = lane & 15;
  int s = laneHi * 8;                                // A-fragment K shift
  const _Float16* ap0 = &sH[(0 * 16 + ln) * 648];
  const _Float16* ap1 = &sH[(1 * 16 + ln) * 648];
  const _Float16* ap2 = &sH[(2 * 16 + ln) * 648];
  const _Float16* ap3 = &sH[(3 * 16 + ln) * 648];

  for (int nt = wave; nt < 63; nt += 8) {
    int v = nt * 16 + ln;
    float b2v = (v < 1000) ? b2[v] : 0.0f;
    v8f acc0 = {}, acc1 = {}, acc2 = {}, acc3 = {};
    const _Float16* bp = W2P + ((size_t)nt * 32 + lane) * 16;
    for (int kt = 0; kt < 20; ++kt) {
      int o = kt * 32 + s;
      v8h blo = *(const v8h*)(bp);
      v8h bhi = *(const v8h*)(bp + 8);
      bp += 63 * 512;                                // next kt plane
      v16h B = SHUF16(blo, bhi);
      v16h A0 = SHUF16(*(const v8h*)(ap0 + o), *(const v8h*)(ap0 + o + 16));
      v16h A1 = SHUF16(*(const v8h*)(ap1 + o), *(const v8h*)(ap1 + o + 16));
      v16h A2 = SHUF16(*(const v8h*)(ap2 + o), *(const v8h*)(ap2 + o + 16));
      v16h A3 = SHUF16(*(const v8h*)(ap3 + o), *(const v8h*)(ap3 + o + 16));
      acc0 = __builtin_amdgcn_wmma_f32_16x16x32_f16(false, A0, false, B,
                                                    (short)0, acc0, false, false);
      acc1 = __builtin_amdgcn_wmma_f32_16x16x32_f16(false, A1, false, B,
                                                    (short)0, acc1, false, false);
      acc2 = __builtin_amdgcn_wmma_f32_16x16x32_f16(false, A2, false, B,
                                                    (short)0, acc2, false, false);
      acc3 = __builtin_amdgcn_wmma_f32_16x16x32_f16(false, A3, false, B,
                                                    (short)0, acc3, false, false);
    }
    if (v < 1000) {
      union { v8f vv; float f[8]; } a0, a1, a2, a3;
      a0.vv = acc0; a1.vv = acc1; a2.vv = acc2; a3.vv = acc3;
      const float* accf[4] = {a0.f, a1.f, a2.f, a3.f};
      for (int mt = 0; mt < 4; ++mt) {
        for (int j = 0; j < 8; ++j) {
          int m = m0 + mt * 16 + j + laneHi * 8;
          int t = m / 400; int rem = m - t * 400;
          int uu = rem >> 3; int nn = rem & 7;
          out[(((size_t)nn * 200 + t) * 50 + uu) * 1000 + v] = accf[mt][j] + b2v;
        }
      }
    }
  }
}

// ---------------------------------------------------------------------------
extern "C" void kernel_launch(void* const* d_in, const int* in_sizes, int n_in,
                              void* d_out, int out_size, void* d_ws, size_t ws_size,
                              hipStream_t stream) {
  const float* enc  = (const float*)d_in[0];   // (200,8,512)
  const float* pred = (const float*)d_in[1];   // (50,8,512)
  const float* W1   = (const float*)d_in[2];   // (640,1024)
  const float* b1   = (const float*)d_in[3];   // (640,)
  const float* W2   = (const float*)d_in[4];   // (1000,640)
  const float* b2   = (const float*)d_in[5];   // (1000,)
  float* out = (float*)d_out;                  // (8,200,50,1000)

  char* ws = (char*)d_ws;
  _Float16* Xh   = (_Float16*)(ws + OFF_XH);
  _Float16* W1PE = (_Float16*)(ws + OFF_W1PE);
  _Float16* W1PP = (_Float16*)(ws + OFF_W1PP);
  _Float16* W2P  = (_Float16*)(ws + OFF_W2P);
  float*    encP  = (float*)(ws + OFF_ENCP);
  float*    predP = (float*)(ws + OFF_PREDP);

  k_cvtX<<<4000, 256, 0, stream>>>(enc, pred, Xh);
  k_packW<<<1280, 256, 0, stream>>>(W1, 1024,   0, 40,  640, 16, W1PE);
  k_packW<<<1280, 256, 0, stream>>>(W1, 1024, 512, 40,  640, 16, W1PP);
  k_packW<<<2520, 256, 0, stream>>>(W2,  640,   0, 63, 1000, 20, W2P);
  k_proj<<<625, 256, 0, stream>>>(Xh, W1PE, W1PP, encP, predP);
  k_joint<<<1250, 256, 0, stream>>>(encP, predP, b1, W2P, b2, out);
}
